// GATModel_24644522345347
// MI455X (gfx1250) — matrix operand order repaired and checked by reading, not compiled, not run on hardware
//
#include <hip/hip_runtime.h>
#include <hip/hip_bf16.h>

// ---------------------------------------------------------------------------
// 2-layer GAT on gfx1250 (MI455X).
//
// Roofline:
//  - GEMMs (21 + 10.5 GFLOP) as bf16 WMMA w/ f32 accum; B panels staged in LDS
//    via the Tensor Data Mover (double-buffered, TENSORcnt-synced) so the 8
//    waves of a block share one 8KB panel fetch instead of 8 redundant ones.
//  - Edge phase: CSR-ize once (counting sort by dst, integer atomics only),
//    then one block per dst does fused max/softmax/aggregate/normalize with
//    register accumulation, zero float atomics. Gather (~0.7GB bf16) stays in
//    the 192MB L2 (h1 is 41MB).
// ---------------------------------------------------------------------------

typedef unsigned short u16;
typedef __attribute__((ext_vector_type(16))) __bf16 v16bf;
typedef __attribute__((ext_vector_type(8)))  float  v8f;
typedef unsigned int uint4v __attribute__((ext_vector_type(4)));
typedef int          int8v  __attribute__((ext_vector_type(8)));
typedef int          int4v  __attribute__((ext_vector_type(4)));

union FragBF16 {
  v16bf v;
  uint4 q[2];
};

__device__ __forceinline__ u16 f2bf(float f) {
  unsigned int u = __float_as_uint(f);
  unsigned int r = 0x7FFFu + ((u >> 16) & 1u);   // round-to-nearest-even
  return (u16)((u + r) >> 16);
}
__device__ __forceinline__ float bf2f(u16 b) {
  return __uint_as_float(((unsigned int)b) << 16);
}

// ---------------------------------------------------------------------------
// TDM: async-copy a 2D bf16 tile (32 K-elems x 128 rows, row stride K elems)
// from global into LDS at byte offset ldsOff (contiguous [row][32] layout).
// Descriptor per CDNA5 ISA (D# groups 0/1; 2/3 zero).
// This toolchain exposes the 6-arg builtin:
//   (uint32x4 g0, int32x8 g1, int32x4 g2, int32x4 g3, int32x8 extra, i32 cpol)
__device__ __forceinline__ void tdm_load_b_panel(const u16* gaddr, unsigned ldsOff,
                                                 int K, int Nc) {
  const unsigned long long ga = (unsigned long long)(uintptr_t)gaddr;
  uint4v g0;
  g0[0] = 1u;                                        // count=1, user mode
  g0[1] = ldsOff;                                    // lds_addr (bytes)
  g0[2] = (unsigned)(ga & 0xffffffffu);              // global_addr[31:0]
  g0[3] = (unsigned)((ga >> 32) & 0x01ffffffu)       // global_addr[56:32]
        | (2u << 30);                                // type=2 ("image")
  int8v g1;
  g1[0] = (1 << 16);                                 // data_size=1 (2 bytes)
  g1[1] = (int)(((unsigned)K & 0xffffu) << 16);      // tensor_dim0[15:0]
  g1[2] = (int)((((unsigned)K >> 16) & 0xffffu)
        | (((unsigned)Nc & 0xffffu) << 16));         // dim0[31:16] | dim1[15:0]
  g1[3] = (int)(((((unsigned)Nc) >> 16) & 0xffffu)
        | (32u << 16));                              // dim1[31:16] | tile_dim0=32
  g1[4] = 128;                                       // tile_dim1=128, tile_dim2=0
  g1[5] = K;                                         // tensor_dim0_stride[31:0]
  g1[6] = 0;
  g1[7] = 0;
  const int4v z4 = {0, 0, 0, 0};                     // groups 2/3: unused (2D)
  const int8v z8 = {0, 0, 0, 0, 0, 0, 0, 0};
  __builtin_amdgcn_tensor_load_to_lds(g0, g1, z4, z4, z8, 0);
}

// ---------------------------------------------------------------------------
__global__ void k_cvt_bf16(const float* __restrict__ in, u16* __restrict__ out,
                           long long n) {
  long long i = (long long)blockIdx.x * blockDim.x + threadIdx.x;
  long long stride = (long long)gridDim.x * blockDim.x;
  for (; i < n; i += stride) out[i] = f2bf(in[i]);
}

// W [K,Nc] f32 row-major -> Wt [Nc,K] bf16 (so B panel rows are contiguous)
__global__ void k_cvt_transpose(const float* __restrict__ W, u16* __restrict__ Wt,
                                int K, int Nc) {
  long long n = (long long)K * Nc;
  long long i = (long long)blockIdx.x * blockDim.x + threadIdx.x;
  long long stride = (long long)gridDim.x * blockDim.x;
  for (; i < n; i += stride) {
    int nc = (int)(i / K);
    int k  = (int)(i % K);
    Wt[i] = f2bf(W[(long long)k * Nc + nc]);
  }
}

__global__ void k_fill_i32(int* __restrict__ p, int v, int n) {
  int i = blockIdx.x * blockDim.x + threadIdx.x;
  int stride = gridDim.x * blockDim.x;
  for (; i < n; i += stride) p[i] = v;
}

// ---------------------------------------------------------------------------
// CSR build: count edges per destination (self-loops appended logically).
__global__ void k_deg_count(const int* __restrict__ ei, int E, int N,
                            int* __restrict__ deg) {
  const int ET = E + N;
  const int stride = gridDim.x * blockDim.x;
  for (int e = blockIdx.x * blockDim.x + threadIdx.x; e < ET; e += stride) {
    const int d = (e < E) ? ei[E + e] : (e - E);
    atomicAdd(&deg[d], 1);
  }
}

// Single-block exclusive scan over deg[0..n) -> rowStart[0..n], cursor copy.
// blockDim must be 1024 (32 waves).
__global__ void k_scan_excl(const int* __restrict__ deg, int* __restrict__ rowStart,
                            int* __restrict__ cursor, int n) {
  __shared__ int wsum[32];
  __shared__ int runningBase;
  const int lane = threadIdx.x & 31;
  const int wid  = threadIdx.x >> 5;
  const int nw   = blockDim.x >> 5;
  if (threadIdx.x == 0) runningBase = 0;
  for (int base = 0; base < n; base += blockDim.x) {
    __syncthreads();
    const int i = base + threadIdx.x;
    const int v = (i < n) ? deg[i] : 0;
    int sv = v;                         // wave inclusive scan
#pragma unroll
    for (int off = 1; off < 32; off <<= 1) {
      int t = __shfl_up(sv, off, 32);
      if (lane >= off) sv += t;
    }
    if (lane == 31) wsum[wid] = sv;
    __syncthreads();
    if (wid == 0) {
      int ws = (lane < nw) ? wsum[lane] : 0;
#pragma unroll
      for (int off = 1; off < 32; off <<= 1) {
        int t = __shfl_up(ws, off, 32);
        if (lane >= off) ws += t;
      }
      if (lane < nw) wsum[lane] = ws;   // inclusive scan of wave sums
    }
    __syncthreads();
    const int waveOff = (wid == 0) ? 0 : wsum[wid - 1];
    const int excl = runningBase + waveOff + (sv - v);
    if (i < n) { rowStart[i] = excl; cursor[i] = excl; }
    __syncthreads();
    if (threadIdx.x == 0) runningBase += wsum[nw - 1];
  }
  __syncthreads();
  if (threadIdx.x == 0) rowStart[n] = runningBase;
}

// Scatter src ids into dst-sorted order.
__global__ void k_scatter(const int* __restrict__ ei, int E, int N,
                          int* __restrict__ cursor, int* __restrict__ srcSorted) {
  const int ET = E + N;
  const int stride = gridDim.x * blockDim.x;
  for (int e = blockIdx.x * blockDim.x + threadIdx.x; e < ET; e += stride) {
    int s, d;
    if (e < E) { s = ei[e]; d = ei[E + e]; } else { s = e - E; d = s; }
    const int pos = atomicAdd(&cursor[d], 1);
    srcSorted[pos] = s;
  }
}

// ---------------------------------------------------------------------------
// bf16 WMMA GEMM with TDM-staged B panels.
//   C[M,Nc](bf16) = A[M,K](bf16) * Bt[Nc,K](bf16)^T, f32 accum.
// Block = 8 waves = 128x128 tile; wave w owns rows [mBase..mBase+16).
// B panel (128 cols x 32 K = 8KB) is TDM-copied to LDS, double-buffered;
// wave 0 issues tensor_load_to_lds for step k+1 while all waves compute k.
// grid: x = Nc/128, y = ceil(M/128). K%64==0.
__global__ void k_wmma_gemm_tdm(const u16* __restrict__ A, const u16* __restrict__ Bt,
                                u16* __restrict__ C, int M, int K, int Nc) {
  __shared__ u16 ldsB[2][128 * 32];          // 2 x 8KB (LDS offset 0 / 8192)
  {
    // The TDM writes LDS behind the compiler's back (the builtin takes a raw
    // LDS byte offset, not a pointer). Escape the buffer address through an
    // INPUT-ONLY asm operand with a memory clobber: the reads below cannot be
    // folded away, yet they keep addrspace(3) provenance -> ds_load_b128
    // (not flat_load).
    const u16* esc = &ldsB[0][0];
    asm volatile("" : : "v"(esc) : "memory");
  }

  const int lane = threadIdx.x & 31;
  const int wave = threadIdx.x >> 5;
  const int r    = lane & 15;                // row (A) / col (B,C) within tile
  const int half = lane >> 4;                // K-half for A/B, M-half for C/D
  const int nBase = blockIdx.x << 7;
  const int mBase = (blockIdx.y * 8 + wave) << 4;

  v8f acc[8];
#pragma unroll
  for (int j = 0; j < 8; ++j)
#pragma unroll
    for (int v = 0; v < 8; ++v) acc[j][v] = 0.0f;

  const int rowA = min(mBase + r, M - 1);    // clamp padded rows in-bounds
  const u16* arow = A + (long long)rowA * K;
  const u16* bpanel = Bt + (long long)nBase * K;

  // prologue: stage k0=0 panel
  if (threadIdx.x < 32) {
    tdm_load_b_panel(bpanel, 0u, K, Nc);
    __builtin_amdgcn_s_wait_tensorcnt(0);
  }
  __syncthreads();

  int cur = 0;
  for (int k0 = 0; k0 < K; k0 += 32) {
    if (threadIdx.x < 32 && (k0 + 32) < K)   // prefetch next panel
      tdm_load_b_panel(bpanel + (k0 + 32), (unsigned)((cur ^ 1) * 8192), K, Nc);

    FragBF16 a;
    // A 16x32: lane (r,half): elems i = A[r][k0+16*half+i]
    a.q[0] = *(const uint4*)(arow + k0 + half * 16);
    a.q[1] = *(const uint4*)(arow + k0 + half * 16 + 8);

    // Compiler-level fence: LDS contents changed since last iteration (TDM).
    asm volatile("" : : : "memory");
#pragma unroll
    for (int j = 0; j < 8; ++j) {
      // LDS panel is [col][32 k]: fragment elems i = B[k0+16*half+i][nBase+col]
      const u16* bp = &ldsB[cur][(j * 16 + r) * 32 + half * 16];
      FragBF16 b;
      b.q[0] = *(const uint4*)(bp);
      b.q[1] = *(const uint4*)(bp + 8);
      acc[j] = __builtin_amdgcn_wmma_f32_16x16x32_bf16(
          false, a.v, false, b.v, (short)0, acc[j], false, false);
    }

    if (threadIdx.x < 32) __builtin_amdgcn_s_wait_tensorcnt(0);
    __syncthreads();                         // next buffer ready; cur consumed
    cur ^= 1;
  }

  // C/D: lane L, VGPR v -> row = mBase + 8*half + v, col = nBase + j*16 + r
#pragma unroll
  for (int j = 0; j < 8; ++j) {
    const int col = nBase + j * 16 + r;
#pragma unroll
    for (int v = 0; v < 8; ++v) {
      const int row = mBase + half * 8 + v;
      if (row < M) C[(long long)row * Nc + col] = f2bf(acc[j][v]);
    }
  }
}

// ---------------------------------------------------------------------------
// a_s[n,h] = sum_c h[n,h,c]*att_src[h,c] ; same for a_d. One wave per (node,head).
__global__ void k_attn_dots(const u16* __restrict__ H,
                            const float* __restrict__ att_s,
                            const float* __restrict__ att_d,
                            float* __restrict__ as, float* __restrict__ ad,
                            int nNodes, int heads) {
  const int lane = threadIdx.x & 31;
  const int total = nNodes * heads;
  const int wstride = (gridDim.x * blockDim.x) >> 5;
  for (int w = (blockIdx.x * blockDim.x + threadIdx.x) >> 5; w < total; w += wstride) {
    const int n = w / heads, h = w % heads;
    const u16* hp = H + (long long)n * heads * 256 + h * 256;
    const float* sp = att_s + h * 256;
    const float* dp = att_d + h * 256;
    float ss = 0.0f, sd = 0.0f;
#pragma unroll
    for (int i = 0; i < 8; ++i) {
      const int c = lane + 32 * i;
      const float hv = bf2f(hp[c]);
      ss += hv * sp[c];
      sd += hv * dp[c];
    }
#pragma unroll
    for (int m = 16; m >= 1; m >>= 1) {
      ss += __shfl_xor(ss, m, 32);
      sd += __shfl_xor(sd, m, 32);
    }
    if (lane == 0) { as[w] = ss; ad[w] = sd; }
  }
}

// ---------------------------------------------------------------------------
// Fused per-destination GAT aggregation (no float atomics):
//   m  = max_e leakyrelu(a_s[src]+a_d[dst])
//   ex = exp(l - m); out = (sum ex*h[src]) / (sum ex) + bias  [; ELU]
// One block per dst node; blockDim = heads*64; thread t handles 4 channels
// c0 = 4t of head h = c0>>8. Self-loops guarantee >=1 edge per node.
// elu!=0 : write bf16 (layer 1);  elu==0 : write f32 (final output).
__global__ void k_gat_agg(const int* __restrict__ rowStart,
                          const int* __restrict__ srcSorted,
                          const float* __restrict__ as,
                          const float* __restrict__ ad,
                          const u16* __restrict__ Hsrc,
                          const float* __restrict__ bias,
                          u16* __restrict__ outB, float* __restrict__ outF,
                          int nNodes, int heads, int elu) {
  const int HC = heads * 256;
  const int c0 = threadIdx.x * 4;
  const int h  = c0 >> 8;
  for (int d = blockIdx.x; d < nNodes; d += gridDim.x) {
    const float adh = ad[d * heads + h];
    const int eb = rowStart[d], ee = rowStart[d + 1];
    float mx = -1e30f;
    for (int e = eb; e < ee; ++e) {
      const int s = srcSorted[e];
      float l = as[s * heads + h] + adh;
      l = (l > 0.0f) ? l : 0.2f * l;                 // leaky relu 0.2
      mx = fmaxf(mx, l);
    }
    float den = 0.0f, a0 = 0.0f, a1 = 0.0f, a2 = 0.0f, a3 = 0.0f;
    for (int e = eb; e < ee; ++e) {
      const int s = srcSorted[e];
      float l = as[s * heads + h] + adh;
      l = (l > 0.0f) ? l : 0.2f * l;
      const float ex = __expf(l - mx);
      den += ex;
      const uint2 pk = *(const uint2*)(Hsrc + (long long)s * HC + c0);
      a0 += ex * bf2f((u16)(pk.x & 0xffffu));
      a1 += ex * bf2f((u16)(pk.x >> 16));
      a2 += ex * bf2f((u16)(pk.y & 0xffffu));
      a3 += ex * bf2f((u16)(pk.y >> 16));
    }
    const float inv = 1.0f / (den + 1e-16f);
    float v0 = a0 * inv + bias[c0 + 0];
    float v1 = a1 * inv + bias[c0 + 1];
    float v2 = a2 * inv + bias[c0 + 2];
    float v3 = a3 * inv + bias[c0 + 3];
    const long long o = (long long)d * HC + c0;
    if (elu) {
      v0 = (v0 > 0.0f) ? v0 : (__expf(v0) - 1.0f);
      v1 = (v1 > 0.0f) ? v1 : (__expf(v1) - 1.0f);
      v2 = (v2 > 0.0f) ? v2 : (__expf(v2) - 1.0f);
      v3 = (v3 > 0.0f) ? v3 : (__expf(v3) - 1.0f);
      outB[o + 0] = f2bf(v0); outB[o + 1] = f2bf(v1);
      outB[o + 2] = f2bf(v2); outB[o + 3] = f2bf(v3);
    } else {
      outF[o + 0] = v0; outF[o + 1] = v1;
      outF[o + 2] = v2; outF[o + 3] = v3;
    }
  }
}

// ---------------------------------------------------------------------------
extern "C" void kernel_launch(void* const* d_in, const int* in_sizes, int n_in,
                              void* d_out, int out_size, void* d_ws, size_t ws_size,
                              hipStream_t stream) {
  const int IN = 512, HEADS = 4, HC1 = 1024, OUT = 256;
  const int N = in_sizes[0] / IN;        // 20000
  const int E = in_sizes[1] / 2;         // 320000
  const int ET = E + N;                  // with self-loops

  const float* x    = (const float*)d_in[0];
  const int*   ei   = (const int*)d_in[1];
  const float* W1   = (const float*)d_in[2];
  const float* at1s = (const float*)d_in[3];
  const float* at1d = (const float*)d_in[4];
  const float* b1   = (const float*)d_in[5];
  const float* W2   = (const float*)d_in[6];
  const float* at2s = (const float*)d_in[7];
  const float* at2d = (const float*)d_in[8];
  const float* b2   = (const float*)d_in[9];
  float* out = (float*)d_out;

  // --- workspace carve-out (aliased where lifetimes allow) ---
  char* ws = (char*)d_ws;
  size_t off = 0;
  auto carve = [&](size_t bytes) {
    void* p = ws + off;
    off = (off + bytes + 255) & ~(size_t)255;
    return p;
  };
  u16*   Xb   = (u16*)  carve((size_t)N * IN * 2);     // x bf16 (later: h2 bf16)
  u16*   W1t  = (u16*)  carve((size_t)HC1 * IN * 2);   // W1^T bf16 [1024,512]
  u16*   W2t  = (u16*)  carve((size_t)OUT * HC1 * 2);  // W2^T bf16 [256,1024]
  u16*   H1b  = (u16*)  carve((size_t)N * HC1 * 2);    // h1 bf16 (later: layer1 out)
  float* as1  = (float*)carve((size_t)N * HEADS * 4);
  float* ad1  = (float*)carve((size_t)N * HEADS * 4);
  float* as2  = (float*)carve((size_t)N * 4);
  float* ad2  = (float*)carve((size_t)N * 4);
  int*   deg  = (int*)  carve((size_t)N * 4);
  int*   rowS = (int*)  carve((size_t)(N + 1) * 4);
  int*   curs = (int*)  carve((size_t)N * 4);
  int*   srcS = (int*)  carve((size_t)ET * 4);
  u16*   H2b  = H1b;   // layer-1 output bf16 (in place over h1, which that k_gat_agg still reads)
  u16*   h2b  = Xb;    // layer-2 h bf16 (x dead by then); N*256*2 <= N*512*2

  // --- 1. conversions ---
  k_cvt_bf16<<<dim3(8192), dim3(256), 0, stream>>>(x, Xb, (long long)N * IN);
  k_cvt_transpose<<<dim3(2048), dim3(256), 0, stream>>>(W1, W1t, IN, HC1);
  k_cvt_transpose<<<dim3(1024), dim3(256), 0, stream>>>(W2, W2t, HC1, OUT);

  // --- 2. CSR build (integer atomics only; reused by both layers) ---
  k_fill_i32<<<dim3(128), dim3(256), 0, stream>>>(deg, 0, N);
  k_deg_count<<<dim3((ET + 255) / 256), dim3(256), 0, stream>>>(ei, E, N, deg);
  k_scan_excl<<<dim3(1), dim3(1024), 0, stream>>>(deg, rowS, curs, N);
  k_scatter<<<dim3((ET + 255) / 256), dim3(256), 0, stream>>>(ei, E, N, curs, srcS);

  const int mBlocks = (N / 16 + 7) / 8;                 // 157 (padded; stores guarded)

  // --- 3. GEMM1: h1 = x @ W1 ---
  k_wmma_gemm_tdm<<<dim3(HC1 / 128, mBlocks), dim3(256), 0, stream>>>(Xb, W1t, H1b, N, IN, HC1);

  // --- 4-5. layer-1 attention + fused aggregate/normalize/ELU ---
  k_attn_dots<<<dim3((N * HEADS + 7) / 8), dim3(256), 0, stream>>>(H1b, at1s, at1d, as1, ad1, N, HEADS);
  k_gat_agg<<<dim3(N), dim3(HEADS * 64), 0, stream>>>(rowS, srcS, as1, ad1, H1b, b1,
                                                      H2b, (float*)nullptr, N, HEADS, 1);

  // --- 6. GEMM2: h2 = H @ W2 ---
  k_wmma_gemm_tdm<<<dim3(OUT / 128, mBlocks), dim3(256), 0, stream>>>(H2b, W2t, h2b, N, HC1, OUT);

  // --- 7. layer-2 attention + fused aggregate -> d_out ---
  k_attn_dots<<<dim3((N + 7) / 8), dim3(256), 0, stream>>>(h2b, at2s, at2d, as2, ad2, N, 1);
  k_gat_agg<<<dim3(N), dim3(64), 0, stream>>>(rowS, srcS, as2, ad2, h2b, b2,
                                              (u16*)nullptr, out, N, 1, 0);
}
